// AttentionPooling_29600914604189
// MI455X (gfx1250) — compile-verified, matrix operand order checked by reading, not run on hardware
//
#include <hip/hip_runtime.h>
#include <hip/hip_bf16.h>

// ---------------------------------------------------------------------------
// AttentionPooling on MI455X (gfx1250), wave32 + WMMA f32_16x16x32_f16.
//
// Pipeline (all on `stream`, scratch in d_ws):
//   1) gather_f16:   emb16[B*S,768]  = (f16) emb_table[input_ids]
//   2) wtrans_f16:   WqT16/WkT16[n,k] = (f16) W[k,n]   (so B-fragments are
//                    contiguous-K row loads, same pattern as A-fragments)
//   3) zero_w:       w[B,H,S] = 0
//   4) proj:         Q,K = emb @ W + b  via v_wmma_f32_16x16x32_f16.
//                    Register-blocked: one wave computes a 64x16 strip of
//                    BOTH Q and K (4 A-fragments reuse the Wq/Wk fragments:
//                    12 b128 loads per 8 WMMA instead of 6 per 2).
//                    Output layout [B,H,S,64] f16 (contiguous dh).
//   5) attn:         per (b,h,qtile16): scores = Q Kt / 8 + bias via WMMA,
//                    16x512 f32 strip in LDS, softmax rows, column sums
//                    atomically accumulated into w  (no [B,H,S,S] in HBM).
//   6) pool:         pooled[b,h,:] = (1/S) * sum_s w[b,h,s] * emb[b,s,:]
// ---------------------------------------------------------------------------

#define Bsz   32
#define Ssz   512
#define Dsz   768
#define Hsz   12
#define DHsz  64
#define TOK   (Bsz * Ssz)          // 16384 token rows
#define NEGB  (-1e9f)
#define MBLK  4                    // M-tiles (of 16 rows) per wave in proj

typedef __attribute__((ext_vector_type(16))) _Float16 v16h;
typedef __attribute__((ext_vector_type(8)))  _Float16 v8h;
typedef __attribute__((ext_vector_type(8)))  float    v8f;

// Load a 16-bit A/B fragment for V_WMMA_*_16X16X32_F16.
// Per ISA layout: lane&15 selects the row (A) / column (B); lane>>4 selects
// which 8-wide K group lives in VGPRs 0-3 vs 4-7:
//   elements 0..7  <- K = kbase + 8*half + (0..7)
//   elements 8..15 <- K = kbase + 8*half + 16 + (0..7)
// Both chunks are 16-byte aligned contiguous loads (global_load_b128).
__device__ __forceinline__ v16h load_frag16(const _Float16* rowptr, int kbase, int half) {
    union { v16h v; v8h h[2]; } u;
    const _Float16* p = rowptr + kbase + 8 * half;
    u.h[0] = *(const v8h*)(p);
    u.h[1] = *(const v8h*)(p + 16);
    return u.v;
}

__device__ __forceinline__ v8f wmma_f16(v16h a, v16h b, v8f c) {
    return __builtin_amdgcn_wmma_f32_16x16x32_f16(
        /*neg_a=*/false, a, /*neg_b=*/false, b,
        /*c_mod=*/(short)0, c, /*reuse_a=*/false, /*reuse_b=*/false);
}

// ---------------------------------------------------------------- kernel 1
__global__ __launch_bounds__(256) void gather_f16_kernel(
    const int* __restrict__ ids, const float* __restrict__ table,
    _Float16* __restrict__ emb16) {
    int t = blockIdx.x;                       // token index [0, TOK)
    int id = ids[t];
    const float* src = table + (size_t)id * Dsz;
    _Float16*    dst = emb16 + (size_t)t  * Dsz;
    for (int d = threadIdx.x; d < Dsz; d += blockDim.x)
        dst[d] = (_Float16)src[d];
}

// ---------------------------------------------------------------- kernel 2
__global__ __launch_bounds__(256) void wtrans_f16_kernel(
    const float* __restrict__ Wq, const float* __restrict__ Wk,
    _Float16* __restrict__ WqT, _Float16* __restrict__ WkT) {
    int n = blockIdx.x;                       // output feature [0, 768)
    for (int k = threadIdx.x; k < Dsz; k += blockDim.x) {
        WqT[(size_t)n * Dsz + k] = (_Float16)Wq[(size_t)k * Dsz + n];
        WkT[(size_t)n * Dsz + k] = (_Float16)Wk[(size_t)k * Dsz + n];
    }
}

// ---------------------------------------------------------------- kernel 3
__global__ __launch_bounds__(256) void zero_kernel(float* __restrict__ p, int n) {
    int i = blockIdx.x * blockDim.x + threadIdx.x;
    if (i < n) p[i] = 0.0f;
}

// ---------------------------------------------------------------- kernel 4
// One wave computes a 64x16 tile of BOTH Q and K (Wq/Wk fragments reused
// across 4 A-fragments). grid = (TOK/(16*MBLK), Dsz/16), block = 32.
__global__ __launch_bounds__(32) void proj_kernel(
    const _Float16* __restrict__ emb16,
    const _Float16* __restrict__ WqT, const _Float16* __restrict__ WkT,
    const float* __restrict__ bq, const float* __restrict__ bk,
    _Float16* __restrict__ Q, _Float16* __restrict__ K) {
    const int mt0 = blockIdx.x * MBLK;        // first M tile of this wave
    const int nt  = blockIdx.y;
    const int lane = threadIdx.x & 31;
    const int half = lane >> 4, l15 = lane & 15;

    const _Float16* arow[MBLK];
    #pragma unroll
    for (int mi = 0; mi < MBLK; ++mi)
        arow[mi] = emb16 + (size_t)((mt0 + mi) * 16 + l15) * Dsz;

    const int n = nt * 16 + l15;              // out feature
    const _Float16* bqrow = WqT + (size_t)n * Dsz;
    const _Float16* bkrow = WkT + (size_t)n * Dsz;

    v8f cq[MBLK] = {};
    v8f ck[MBLK] = {};
    for (int kb = 0; kb < Dsz; kb += 32) {
        v16h wq = load_frag16(bqrow, kb, half);
        v16h wk = load_frag16(bkrow, kb, half);
        #pragma unroll
        for (int mi = 0; mi < MBLK; ++mi) {
            v16h a = load_frag16(arow[mi], kb, half);
            cq[mi] = wmma_f16(a, wq, cq[mi]);
            ck[mi] = wmma_f16(a, wk, ck[mi]);
        }
    }

    const float bqv = bq[n];
    const float bkv = bk[n];
    const int h  = n / DHsz;
    const int dh = n % DHsz;
    // C/D layout: VGPR r holds (M = r + 8*half, N = l15)
    #pragma unroll
    for (int mi = 0; mi < MBLK; ++mi) {
        #pragma unroll
        for (int r = 0; r < 8; ++r) {
            int t = (mt0 + mi) * 16 + r + 8 * half;   // token
            int b = t / Ssz, s = t % Ssz;
            size_t o = ((((size_t)b * Hsz + h) * Ssz) + s) * DHsz + dh;
            Q[o] = (_Float16)(cq[mi][r] + bqv);
            K[o] = (_Float16)(ck[mi][r] + bkv);
        }
    }
}

// ---------------------------------------------------------------- kernel 5
// grid = (S/16, H, B), block = 32. Fused scores->softmax->column-sum.
__global__ __launch_bounds__(32) void attn_kernel(
    const _Float16* __restrict__ Q, const _Float16* __restrict__ K,
    const int* __restrict__ mask, float* __restrict__ w) {
    __shared__ float sc[16][Ssz];            // 32 KB score strip
    __shared__ float rmax[16], rinv[16];

    const int qt = blockIdx.x, h = blockIdx.y, b = blockIdx.z;
    const int lane = threadIdx.x & 31;
    const int half = lane >> 4, l15 = lane & 15;

    const size_t headoff = ((size_t)b * Hsz + h) * Ssz;
    const _Float16* qrow = Q + (headoff + qt * 16 + l15) * DHsz;
    v16h a0 = load_frag16(qrow, 0,  half);   // dh 0..31
    v16h a1 = load_frag16(qrow, 32, half);   // dh 32..63

    const _Float16* kbase = K + headoff * DHsz;
    const int* mrow = mask + (size_t)b * Ssz;

    for (int kt = 0; kt < Ssz / 16; ++kt) {
        const int key = kt * 16 + l15;
        const _Float16* krow = kbase + (size_t)key * DHsz;
        // keep the K stream ahead of the WMMA chain (global_prefetch path)
        if (kt + 1 < Ssz / 16)
            __builtin_prefetch(krow + 16 * DHsz, 0, 1);
        v16h b0 = load_frag16(krow, 0,  half);
        v16h b1 = load_frag16(krow, 32, half);
        v8f c = {};
        c = wmma_f16(a0, b0, c);
        c = wmma_f16(a1, b1, c);
        const float bias = (1.0f - (float)mrow[key]) * NEGB;
        #pragma unroll
        for (int r = 0; r < 8; ++r)
            sc[r + 8 * half][key] = c[r] * 0.125f + bias;   // 1/sqrt(64)
    }
    __syncthreads();

    // pass 1: per-row max and 1/denominator (lanes 0..15, one row each)
    if (lane < 16) {
        float mx = -3.402823e38f;
        for (int k2 = 0; k2 < Ssz; ++k2) mx = fmaxf(mx, sc[lane][k2]);
        float den = 0.0f;
        for (int k2 = 0; k2 < Ssz; ++k2) den += __expf(sc[lane][k2] - mx);
        rmax[lane] = mx;
        rinv[lane] = 1.0f / den;
    }
    __syncthreads();

    // pass 2: column sums over the 16 query rows -> atomic accumulate into w
    float* wrow = w + headoff;
    for (int k2 = lane; k2 < Ssz; k2 += 32) {
        float acc = 0.0f;
        #pragma unroll
        for (int m = 0; m < 16; ++m)
            acc += __expf(sc[m][k2] - rmax[m]) * rinv[m];
        atomicAdd(&wrow[k2], acc);
    }
}

// ---------------------------------------------------------------- kernel 6
// grid = B, block = 256. pooled[b,h,d] = (1/S) sum_s w[b,h,s] * emb[b,s,d]
__global__ __launch_bounds__(256) void pool_kernel(
    const _Float16* __restrict__ emb16, const float* __restrict__ w,
    float* __restrict__ out) {
    __shared__ float wl[Hsz * Ssz];          // 24 KB
    const int b = blockIdx.x;
    for (int i = threadIdx.x; i < Hsz * Ssz; i += blockDim.x)
        wl[i] = w[(size_t)b * Hsz * Ssz + i] * (1.0f / (float)Ssz);
    __syncthreads();

    const _Float16* eb = emb16 + (size_t)b * Ssz * Dsz;
    for (int d = threadIdx.x; d < Dsz; d += blockDim.x) {
        float acc[Hsz] = {};
        for (int s = 0; s < Ssz; ++s) {
            float e = (float)eb[(size_t)s * Dsz + d];
            #pragma unroll
            for (int h = 0; h < Hsz; ++h) acc[h] += wl[h * Ssz + s] * e;
        }
        #pragma unroll
        for (int h = 0; h < Hsz; ++h)
            out[((size_t)b * Hsz + h) * Dsz + d] = acc[h];
    }
}

// ---------------------------------------------------------------------------
extern "C" void kernel_launch(void* const* d_in, const int* in_sizes, int n_in,
                              void* d_out, int out_size, void* d_ws, size_t ws_size,
                              hipStream_t stream) {
    const int*   input_ids = (const int*)  d_in[0];
    const int*   attn_mask = (const int*)  d_in[1];
    const float* emb_table = (const float*)d_in[2];
    const float* Wq        = (const float*)d_in[3];
    const float* bq        = (const float*)d_in[4];
    const float* Wk        = (const float*)d_in[5];
    const float* bk        = (const float*)d_in[6];
    float*       out       = (float*)d_out;

    // Scratch layout (256B aligned slices of d_ws)
    auto align256 = [](size_t x) { return (x + 255) & ~(size_t)255; };
    char* ws = (char*)d_ws;
    size_t off = 0;
    _Float16* emb16 = (_Float16*)(ws + off); off += align256((size_t)TOK * Dsz * 2);       // 24 MB
    _Float16* WqT   = (_Float16*)(ws + off); off += align256((size_t)Dsz * Dsz * 2);       // 1.125 MB
    _Float16* WkT   = (_Float16*)(ws + off); off += align256((size_t)Dsz * Dsz * 2);
    _Float16* Qb    = (_Float16*)(ws + off); off += align256((size_t)TOK * Dsz * 2);       // 24 MB
    _Float16* Kb    = (_Float16*)(ws + off); off += align256((size_t)TOK * Dsz * 2);       // 24 MB
    float*    wacc  = (float*)   (ws + off); off += align256((size_t)Bsz * Hsz * Ssz * 4); // 768 KB

    // 1) gather + f32->f16
    gather_f16_kernel<<<TOK, 256, 0, stream>>>(input_ids, emb_table, emb16);
    // 2) weight transpose + f32->f16
    wtrans_f16_kernel<<<Dsz, 256, 0, stream>>>(Wq, Wk, WqT, WkT);
    // 3) zero the w accumulator (fresh every call; atomics accumulate into it)
    {
        int n = Bsz * Hsz * Ssz;
        zero_kernel<<<(n + 255) / 256, 256, 0, stream>>>(wacc, n);
    }
    // 4) Q/K projection GEMMs via WMMA (64x16 per wave, weight-fragment reuse)
    {
        dim3 grid(TOK / (16 * MBLK), Dsz / 16);
        proj_kernel<<<grid, 32, 0, stream>>>(emb16, WqT, WkT, bq, bk, Qb, Kb);
    }
    // 5) fused attention scores + softmax + query-mean accumulation
    {
        dim3 grid(Ssz / 16, Hsz, Bsz);
        attn_kernel<<<grid, 32, 0, stream>>>(Qb, Kb, attn_mask, wacc);
    }
    // 6) pooled output
    pool_kernel<<<Bsz, 256, 0, stream>>>(emb16, wacc, out);
}
